// ParticleNet_12300786335757
// MI455X (gfx1250) — matrix operand order repaired and checked by reading, not compiled, not run on hardware
//
#include <hip/hip_runtime.h>

typedef __attribute__((ext_vector_type(16))) _Float16 v16h;
typedef __attribute__((ext_vector_type(8)))  _Float16 v8h;
typedef __attribute__((ext_vector_type(8)))  float    v8f;

#define NF   8192
#define NB   4096
#define KN   48
#define RADI 0.1125f
#define DTs  0.02f

// ---------------------------------------------------------------- helpers
static __device__ __forceinline__ unsigned pk2h(float a, float b) {
    _Float16 ha = (_Float16)a, hb = (_Float16)b;
    unsigned short ua = __builtin_bit_cast(unsigned short, ha);
    unsigned short ub = __builtin_bit_cast(unsigned short, hb);
    return (unsigned)ua | ((unsigned)ub << 16);
}

// ---------------------------------------------------------------- prep
__global__ void prep_kernel(const float* __restrict__ pos, const float* __restrict__ vel,
                            const int* __restrict__ nbrF,
                            float* __restrict__ pos_new, float* __restrict__ feats0,
                            float* __restrict__ dcount) {
    int n = blockIdx.x * blockDim.x + threadIdx.x;
    if (n >= NF) return;
    float vx = vel[n*3+0], vy = vel[n*3+1], vz = vel[n*3+2];
    float vnx = vx, vny = vy - 9.81f * DTs, vnz = vz;
    float pnx = pos[n*3+0] + (vx+vnx)*(0.5f*DTs);
    float pny = pos[n*3+1] + (vy+vny)*(0.5f*DTs);
    float pnz = pos[n*3+2] + (vz+vnz)*(0.5f*DTs);
    pos_new[n*3+0]=pnx; pos_new[n*3+1]=pny; pos_new[n*3+2]=pnz;
    feats0[n*16+0]=1.f; feats0[n*16+1]=vnx; feats0[n*16+2]=vny; feats0[n*16+3]=vnz;
    #pragma unroll
    for (int c = 4; c < 16; ++c) feats0[n*16+c] = 0.f;
    int cnt = 0;
    for (int k = 0; k < KN; ++k) cnt += (nbrF[n*KN+k] >= 0) ? 1 : 0;
    dcount[n] = (float)cnt;
}

__global__ void pad_box_kernel(const float* __restrict__ bf, float* __restrict__ out) {
    int i = blockIdx.x * blockDim.x + threadIdx.x;
    if (i >= NB*16) return;
    int b = i >> 4, c = i & 15;
    out[i] = (c < 3) ? bf[b*3+c] : 0.f;
}

// ---------------------------------------------------------------- geometry
// per (n,k): compact 24B record {idx, tapbase, win, fx, fy, fz}
__global__ void geom_kernel(const float* __restrict__ pout, const float* __restrict__ pin,
                            const int* __restrict__ nbr, unsigned* __restrict__ go,
                            int total) {
    int gid = blockIdx.x * blockDim.x + threadIdx.x;
    if (gid >= total) return;
    int n = gid / KN;
    int idx = nbr[gid];
    unsigned* g = go + (size_t)gid * 6;
    if (idx < 0) {
        g[0]=0u; g[1]=0u; g[2]=0u; g[3]=0u; g[4]=0u; g[5]=0u;
        return;
    }
    const float inv_r = 1.0f / RADI;
    float x = (pin[idx*3+0] - pout[n*3+0]) * inv_r;
    float y = (pin[idx*3+1] - pout[n*3+1]) * inv_r;
    float z = (pin[idx*3+2] - pout[n*3+2]) * inv_r;
    float d2 = x*x + y*y + z*z;
    float wb = 1.f - d2;
    float win = fminf(fmaxf(wb*wb*wb, 0.f), 1.f);
    // ball -> cube (volume preserving)
    float rxy2 = x*x + y*y;
    float rho  = sqrtf(rxy2 + z*z + 1e-8f);
    float rxy  = sqrtf(rxy2 + 1e-8f);
    bool  pole = rxy2 <= 1.25f * z * z;
    float s_pole = sqrtf(3.f * rho / (rho + fabsf(z)));
    float s_side = rho / rxy;
    float cx = pole ? x * s_pole : x * s_side;
    float cy = pole ? y * s_pole : y * s_side;
    float sgnz = (z > 0.f) ? 1.f : ((z < 0.f) ? -1.f : 0.f);
    float cz = pole ? sgnz * rho : 1.5f * z;
    float r  = sqrtf(cx*cx + cy*cy + 1e-8f);
    bool  cond = fabsf(cx) >= fabsf(cy);
    float sx = (cx >= 0.f) ? 1.f : -1.f;
    float sy = (cy >= 0.f) ? 1.f : -1.f;
    float scx = (fabsf(cx) > 1e-8f) ? cx : 1.f;
    float scy = (fabsf(cy) > 1e-8f) ? cy : 1.f;
    const float c4pi = 1.2732395447351628f;
    float u = cond ? sx * r : sy * c4pi * r * atanf(cx / scy);
    float v = cond ? sx * c4pi * r * atanf(cy / scx) : sy * r;
    bool tiny = rho < 1e-6f;
    float cu = tiny ? 0.f : u, cv = tiny ? 0.f : v, cw = tiny ? 0.f : cz;
    // trilinear taps
    float t0 = fminf(fmaxf((cu*0.5f + 0.5f)*3.f, 0.f), 3.f);
    float t1 = fminf(fmaxf((cv*0.5f + 0.5f)*3.f, 0.f), 3.f);
    float t2 = fminf(fmaxf((cw*0.5f + 0.5f)*3.f, 0.f), 3.f);
    int i0 = min(max((int)floorf(t0), 0), 2);
    int i1 = min(max((int)floorf(t1), 0), 2);
    int i2 = min(max((int)floorf(t2), 0), 2);
    float fx = t0 - (float)i0, fy = t1 - (float)i1, fz = t2 - (float)i2;
    int tb = (i0*4 + i1)*4 + i2;
    g[0] = (unsigned)idx;
    g[1] = (unsigned)tb;
    g[2] = __builtin_bit_cast(unsigned, win);
    g[3] = __builtin_bit_cast(unsigned, fx);
    g[4] = __builtin_bit_cast(unsigned, fy);
    g[5] = __builtin_bit_cast(unsigned, fz);
}

// ---------------------------------------------------------------- filter pack
// WMMA B layout: N = lane%16, K = table order (8*(lane/16) + j%8 + 16*(j/8)).
// Conv region: [w][cc][s][lane][j], K-row -> (t = 2s + j/8, c = 16cc + 8*(lane/16) + j%8)
// Dense region appended: [w][s][lane][j], K-row -> c = 32s + 16*(j/8) + 8*(lane/16) + j%8.
__global__ void pack_kernel(const float* __restrict__ convW, const float* __restrict__ convB,
                            const float* __restrict__ denseW, const float* __restrict__ denseB,
                            _Float16* __restrict__ pk, float* __restrict__ biasOut,
                            int C16, int Cd32, int D16, int Ca, int Da) {
    int per = (C16*32 + Cd32) * 512;
    int total = D16 * per;
    int nb = D16 * 16;
    for (int i = blockIdx.x*blockDim.x + threadIdx.x; i < total + nb; i += gridDim.x*blockDim.x) {
        if (i < total) {
            int w = i / per, r = i % per;
            int convsz = C16 * 32 * 512;
            float v = 0.f;
            if (r < convsz) {
                int cc = r / (32*512); int r2 = r % (32*512);
                int s = r2 / 512; int li = r2 % 512;
                int l = li >> 4, j = li & 15;
                int t = 2*s + (j >> 3);
                int c = cc*16 + (l >> 4)*8 + (j & 7);
                int d = w*16 + (l & 15);
                if (c < Ca && d < Da) v = convW[((size_t)t*Ca + c)*Da + d];
            } else {
                int r2 = r - convsz;
                int s = r2 / 512; int li = r2 % 512;
                int l = li >> 4, j = li & 15;
                int c = s*32 + (j >> 3)*16 + (l >> 4)*8 + (j & 7);
                int d = w*16 + (l & 15);
                if (c < Ca && d < Da) v = denseW[(size_t)c*Da + d];
            }
            pk[i] = (_Float16)v;
        } else {
            int d = i - total;
            float b = 0.f;
            if (d < Da) {
                if (convB)  b += convB[d];
                if (denseB) b += denseB[d];
            }
            biasOut[d] = b;
        }
    }
}

// ---------------------------------------------------------------- cconv
// Block = 16 query particles, 256 threads (8 waves; 4 run WMMA, all scatter).
// Geometry records for the tile staged in LDS once; per 16-channel chunk the
// neighbor feats are scattered into an f16 LDS patch A[16][64][16] with
// ds_pk_add_f16, then 32 K-steps of v_wmma_f32_16x16x32_f16 per output wave.
// Dense branch fused as extra WMMA K-steps from an f16 copy of tile feats.
template<int C16, int Cd32, int D16, bool RELU, bool RES, bool FINAL>
__global__ __launch_bounds__(256)
void cconv_kernel(const unsigned* __restrict__ geom,
                  const float* __restrict__ feat, int FS,
                  const _Float16* __restrict__ pk,
                  const float* __restrict__ bias,
                  const float* __restrict__ resid,
                  float* __restrict__ out, int OS, int ooff,
                  const float* __restrict__ pos_new,
                  const float* __restrict__ pos,
                  float* __restrict__ dout) {
    constexpr int CPD = (Cd32 > 0) ? Cd32*32 : 1;
    __shared__ __align__(16) unsigned  Abuf[(C16 > 0) ? 16*64*8 : 4];   // A as half2
    __shared__ __align__(16) unsigned  Gbuf[(C16 > 0) ? 16*KN*6 : 4];   // geom records
    __shared__ __align__(16) _Float16 Hbuf[(Cd32 > 0) ? 16*CPD : 8];

    const int tid  = threadIdx.x;
    const int lane = tid & 31;
    const int wv   = tid >> 5;
    const int tile = blockIdx.x;

    if constexpr (Cd32 > 0) {
        for (int i = tid; i < 16*CPD; i += 256) {
            int m = i / CPD, c = i % CPD;
            float v = (c < FS) ? feat[(size_t)(tile*16 + m)*FS + c] : 0.f;
            if (RELU) v = (v > 0.f) ? v : 0.f;
            Hbuf[i] = (_Float16)v;
        }
    }
    if constexpr (C16 > 0) {
        const unsigned* gsrc = geom + (size_t)tile*16*KN*6;
        for (int i = tid; i < 16*KN*6; i += 256) Gbuf[i] = gsrc[i];
    }

    v8f acc = {0.f,0.f,0.f,0.f,0.f,0.f,0.f,0.f};
    const int perW = (C16*32 + Cd32) * 512;

    if constexpr (C16 > 0) {
        const int q  = tid & 7;    // channel pair group (2 channels)
        const int pl = tid >> 3;   // 32 pairs in flight
        for (int cc = 0; cc < C16; ++cc) {
            for (int i = tid; i < 16*64*8; i += 256) Abuf[i] = 0u;
            // warm L2/L0 with this chunk's B tile while we scatter
            if (wv < D16) {
                const char* pb = (const char*)(pk + (size_t)wv*perW + (size_t)cc*32*512);
                #pragma unroll
                for (int pf = 0; pf < 8; ++pf)
                    __builtin_prefetch(pb + lane*1024 + pf*128, 0, 3);
            }
            __syncthreads();
            // ---- scatter (geometry from LDS, corner weights recomputed)
            for (int p0 = 0; p0 < 16*KN; p0 += 32) {
                int p = p0 + pl;
                const unsigned* g = Gbuf + p*6;
                int   idx = (int)g[0];
                int   tb  = (int)g[1];
                float win = __builtin_bit_cast(float, g[2]);
                float fx  = __builtin_bit_cast(float, g[3]);
                float fy  = __builtin_bit_cast(float, g[4]);
                float fz  = __builtin_bit_cast(float, g[5]);
                float gx0 = win*(1.f-fx), gx1 = win*fx;
                int m = p / KN;
                float f0 = feat[(size_t)idx*FS + cc*16 + 2*q + 0];
                float f1 = feat[(size_t)idx*FS + cc*16 + 2*q + 1];
                if (RELU) { f0 = (f0 > 0.f) ? f0 : 0.f; f1 = (f1 > 0.f) ? f1 : 0.f; }
                constexpr int toff_[8] = {0,1,4,5,16,17,20,21};
                #pragma unroll
                for (int c = 0; c < 8; ++c) {
                    int dx = (c>>2)&1, dy = (c>>1)&1, dz = c&1;
                    float wcv = (dx ? gx1 : gx0) * (dy ? fy : 1.f-fy) * (dz ? fz : 1.f-fz);
                    unsigned val = pk2h(wcv*f0, wcv*f1);
                    int tap = tb + toff_[c];
                    unsigned addr = (unsigned)(unsigned long long)(&Abuf[(m*64 + tap)*8 + q]);
                    asm volatile("ds_pk_add_f16 %0, %1" :: "v"(addr), "v"(val) : "memory");
                }
            }
            asm volatile("s_wait_dscnt 0x0" ::: "memory");
            __syncthreads();
            // ---- WMMA over this chunk: 32 K-steps of (2 taps x 16 ch)
            if (wv < D16) {
                const _Float16* Ah = (const _Float16*)Abuf;
                const _Float16* pkw = pk + (size_t)wv * perW;
                const int M  = lane & 15;
                const int hi = (lane >> 4) & 1;
                #pragma unroll
                for (int s = 0; s < 32; ++s) {
                    v8h r0 = *(const v8h*)(Ah + (M*64 + 2*s + 0)*16 + hi*8);
                    v8h r1 = *(const v8h*)(Ah + (M*64 + 2*s + 1)*16 + hi*8);
                    v16h a = __builtin_shufflevector(r0, r1, 0,1,2,3,4,5,6,7,8,9,10,11,12,13,14,15);
                    v16h b = *(const v16h*)(pkw + ((size_t)(cc*32 + s))*512 + lane*16);
                    acc = __builtin_amdgcn_wmma_f32_16x16x32_f16(false, a, false, b,
                                                                 (short)0, acc, false, false);
                }
            }
            __syncthreads();
        }
    }

    if constexpr (Cd32 > 0) {
        if constexpr (C16 == 0) __syncthreads();   // Hbuf ready (conv path already synced)
        if (wv < D16) {
            const _Float16* pkw = pk + (size_t)wv * perW + (size_t)C16*32*512;
            const int M  = lane & 15;
            const int hi = (lane >> 4) & 1;
            #pragma unroll
            for (int s = 0; s < Cd32; ++s) {
                v8h r0 = *(const v8h*)(Hbuf + M*CPD + s*32 + hi*8);
                v8h r1 = *(const v8h*)(Hbuf + M*CPD + s*32 + hi*8 + 16);
                v16h a = __builtin_shufflevector(r0, r1, 0,1,2,3,4,5,6,7,8,9,10,11,12,13,14,15);
                v16h b = *(const v16h*)(pkw + (size_t)s*512 + lane*16);
                acc = __builtin_amdgcn_wmma_f32_16x16x32_f16(false, a, false, b,
                                                             (short)0, acc, false, false);
            }
        }
    }

    // ---- epilogue (C/D layout: VGPR r, lane l -> M = r + 8*(l>=16), N = l%16)
    if (wv < D16) {
        const int nn = lane & 15;
        const int hi = lane >> 4;
        const int d  = wv*16 + nn;
        const float bd = bias[d];
        #pragma unroll
        for (int r = 0; r < 8; ++r) {
            int m = r + 8*hi;
            int gidx = tile*16 + m;
            float v = acc[r] + bd;
            if constexpr (RES) v += resid[(size_t)gidx*FS + d];
            if constexpr (FINAL) {
                if (d < 3) {
                    float corr = v * (1.0f/128.0f);
                    float pc = pos_new[gidx*3 + d] + corr;
                    dout[gidx*3 + d] = pc;
                    dout[NF*3 + gidx*3 + d] = (pc - pos[gidx*3 + d]) * (1.0f/DTs);
                }
            } else {
                out[(size_t)gidx*OS + ooff + d] = v;
            }
        }
    }
}

// ---------------------------------------------------------------- launch
extern "C" void kernel_launch(void* const* d_in, const int* in_sizes, int n_in,
                              void* d_out, int out_size, void* d_ws, size_t ws_size,
                              hipStream_t stream) {
    (void)in_sizes; (void)n_in; (void)out_size; (void)ws_size;
    const float* pos   = (const float*)d_in[0];
    const float* vel   = (const float*)d_in[1];
    const float* box   = (const float*)d_in[2];
    const float* bfeat = (const float*)d_in[3];
    const int* nbrF    = (const int*)d_in[4];
    const int* nbrB    = (const int*)d_in[5];
    const float* cf0_w = (const float*)d_in[6],  *cf0_b = (const float*)d_in[7];
    const float* co0_w = (const float*)d_in[8],  *co0_b = (const float*)d_in[9];
    const float* d0_w  = (const float*)d_in[10], *d0_b  = (const float*)d_in[11];
    const float* c1_w  = (const float*)d_in[12], *c1_b  = (const float*)d_in[13];
    const float* d1_w  = (const float*)d_in[14], *d1_b  = (const float*)d_in[15];
    const float* c2_w  = (const float*)d_in[16], *c2_b  = (const float*)d_in[17];
    const float* d2_w  = (const float*)d_in[18], *d2_b  = (const float*)d_in[19];
    const float* c3_w  = (const float*)d_in[20], *c3_b  = (const float*)d_in[21];
    const float* d3_w  = (const float*)d_in[22], *d3_b  = (const float*)d_in[23];
    float* dout = (float*)d_out;

    // workspace carve-up (floats)
    float* wsf = (float*)d_ws;
    size_t o = 0;
    float* pos_new = wsf + o; o += (size_t)NF*3;
    float* feats0  = wsf + o; o += (size_t)NF*16;
    float* bpad    = wsf + o; o += (size_t)NB*16;
    unsigned* geomF = (unsigned*)(wsf + o); o += (size_t)NF*KN*6;
    unsigned* geomB = (unsigned*)(wsf + o); o += (size_t)NF*KN*6;
    float* x0 = wsf + o; o += (size_t)NF*96;
    float* x1 = wsf + o; o += (size_t)NF*64;
    float* x2 = wsf + o; o += (size_t)NF*64;
    o = (o + 15) & ~(size_t)15;                    // 64B align for packed f16
    _Float16* pkbase = (_Float16*)(wsf + o);
    size_t oh = 0;
    _Float16* pk_co0 = pkbase + oh; oh += (size_t)2*(1*32+0)*512;
    _Float16* pk_cf0 = pkbase + oh; oh += (size_t)2*(1*32+0)*512;
    _Float16* pk_d0  = pkbase + oh; oh += (size_t)2*(0*32+1)*512;
    _Float16* pk_c1  = pkbase + oh; oh += (size_t)4*(6*32+3)*512;
    _Float16* pk_c2  = pkbase + oh; oh += (size_t)4*(4*32+2)*512;
    _Float16* pk_c3  = pkbase + oh; oh += (size_t)1*(4*32+2)*512;
    oh = (oh + 31) & ~(size_t)31;
    float* bb = (float*)(pkbase + oh);
    float* b_co0 = bb;       float* b_cf0 = bb + 32;  float* b_d0 = bb + 64;
    float* b_c1  = bb + 96;  float* b_c2  = bb + 160; float* b_c3 = bb + 224;

    // 1) prep + padding
    prep_kernel<<<(NF+255)/256, 256, 0, stream>>>(pos, vel, nbrF, pos_new, feats0,
                                                  dout + (size_t)NF*6);
    pad_box_kernel<<<(NB*16+255)/256, 256, 0, stream>>>(bfeat, bpad);

    // 2) geometry (fluid reused by 4 convs; box once)
    int tp = NF*KN;
    geom_kernel<<<(tp+255)/256, 256, 0, stream>>>(pos_new, pos_new, nbrF, geomF, tp);
    geom_kernel<<<(tp+255)/256, 256, 0, stream>>>(pos_new, box,     nbrB, geomB, tp);

    // 3) pack filters into WMMA-B layout (+ combined biases)
    pack_kernel<<<256, 256, 0, stream>>>(co0_w, co0_b, nullptr, nullptr, pk_co0, b_co0, 1,0,2, 3, 32);
    pack_kernel<<<256, 256, 0, stream>>>(cf0_w, cf0_b, nullptr, nullptr, pk_cf0, b_cf0, 1,0,2, 4, 32);
    pack_kernel<<<64,  256, 0, stream>>>(nullptr, nullptr, d0_w, d0_b,  pk_d0,  b_d0,  0,1,2, 4, 32);
    pack_kernel<<<1024,256, 0, stream>>>(c1_w, c1_b, d1_w, d1_b,        pk_c1,  b_c1,  6,3,4, 96,64);
    pack_kernel<<<1024,256, 0, stream>>>(c2_w, c2_b, d2_w, d2_b,        pk_c2,  b_c2,  4,2,4, 64,64);
    pack_kernel<<<512, 256, 0, stream>>>(c3_w, c3_b, d3_w, d3_b,        pk_c3,  b_c3,  4,2,1, 64,3);

    dim3 g(NF/16), b(256);
    // 4) layer 0: x0 = [ a_obs | a_conv | a_dense ]
    cconv_kernel<1,0,2,false,false,false><<<g,b,0,stream>>>(geomB, bpad,   16, pk_co0, b_co0,
        nullptr, x0, 96, 0,  nullptr, nullptr, nullptr);
    cconv_kernel<1,0,2,false,false,false><<<g,b,0,stream>>>(geomF, feats0, 16, pk_cf0, b_cf0,
        nullptr, x0, 96, 32, nullptr, nullptr, nullptr);
    cconv_kernel<0,1,2,false,false,false><<<g,b,0,stream>>>(geomF, feats0, 16, pk_d0,  b_d0,
        nullptr, x0, 96, 64, nullptr, nullptr, nullptr);
    // 5) layer 1: x1 = conv(relu(x0)) + dense(relu(x0))
    cconv_kernel<6,3,4,true,false,false><<<g,b,0,stream>>>(geomF, x0, 96, pk_c1, b_c1,
        nullptr, x1, 64, 0, nullptr, nullptr, nullptr);
    // 6) layer 2: x2 = conv(relu(x1)) + dense(relu(x1)) + x1
    cconv_kernel<4,2,4,true,true,false><<<g,b,0,stream>>>(geomF, x1, 64, pk_c2, b_c2,
        x1, x2, 64, 0, nullptr, nullptr, nullptr);
    // 7) layer 3 (final): write pos_new_corrected + vel_new_corrected
    cconv_kernel<4,2,1,true,false,true><<<g,b,0,stream>>>(geomF, x2, 64, pk_c3, b_c3,
        nullptr, nullptr, 0, 0, pos_new, pos, dout);
}
